// Critic_ObstacleEncoder_80169859547829
// MI455X (gfx1250) — matrix operand order repaired and checked by reading, not compiled, not run on hardware
//
#include <hip/hip_runtime.h>
#include <hip/hip_bf16.h>

// ---------------------------------------------------------------------------
// MI455X (gfx1250) fused implementation.
//   Kernel 0 (k_prep):    f32 weights -> transposed f16 weights in d_ws.
//   Kernel 1 (k_encoder): per-batch (16 landmark rows) fused enc-MLP, val-MLP,
//                         softmax attention -> obstacle_mean_embed (out region 2).
//   Kernel 2 (k_tail):    per-16-row fused vals-MLP, layernorm, grouped
//                         8-head attention, x4 agent tile -> out region 1.
// All matmuls use v_wmma_f32_16x16x32_f16 (wave32, 16x16 tiles, f32 accum).
// ---------------------------------------------------------------------------

typedef _Float16 h16 __attribute__((ext_vector_type(16)));
typedef _Float16 h8  __attribute__((ext_vector_type(8)));
typedef float    f8  __attribute__((ext_vector_type(8)));

#define BATCH 16384
#define HID   256

__device__ __forceinline__ f8 wmma_f16(h16 a, h16 b, f8 c) {
  // D = A(16x32 f16) x B(32x16 f16) + C(16x16 f32)
  return __builtin_amdgcn_wmma_f32_16x16x32_f16(
      /*neg_a=*/false, a, /*neg_b=*/false, b,
      /*c_mod=*/(short)0, c, /*reuse_a=*/false, /*reuse_b=*/false);
}

// A fragment: ISA 7.12.2 16-bit A 16x32. Lane half h: VGPR0-3 hold K=h*8+0..7,
// VGPR4-7 hold K=16+h*8+0..7 -> two contiguous 8-half (16B) LDS loads.
__device__ __forceinline__ h16 load_a_lds(const _Float16* row, int k0, int h) {
  const h8 lo = *(const h8*)(row + k0 + h * 8);
  const h8 hi = *(const h8*)(row + k0 + 16 + h * 8);
  return __builtin_shufflevector(lo, hi, 0,1,2,3,4,5,6,7,8,9,10,11,12,13,14,15);
}

// B fragment: 16-bit B 32x16, same K order as A (lane half h: VGPR0-3 hold
// K=h*8+0..7, VGPR4-7 hold K=16+h*8+0..7). With W stored transposed ([N][K]
// f16) this is two contiguous 8-half (16B) global loads served from L2.
__device__ __forceinline__ h16 load_b_gl(const _Float16* wt_row, int k0, int h) {
  const h8 lo = *(const h8*)(wt_row + k0 + h * 8);
  const h8 hi = *(const h8*)(wt_row + k0 + 16 + h * 8);
  return __builtin_shufflevector(lo, hi, 0,1,2,3,4,5,6,7,8,9,10,11,12,13,14,15);
}

// Fast tanh: 1 - 2/(exp(2x)+1) with raw v_rcp_f32 (no IEEE divide expansion).
// Argument clamped so exp never overflows; 2 TRANS + 3 VALU ops, co-executes
// with the XDL WMMA pipe on CDNA5.
__device__ __forceinline__ float ftanh(float x) {
  x = fminf(fmaxf(x, -10.f), 10.f);
  float e = __expf(2.f * x);
  return 1.f - 2.f * __builtin_amdgcn_rcpf(e + 1.f);
}

// ---------------------------------------------------------------------------
// Kernel 0: transpose+convert f32 [K][N] -> f16 [N][Kpad] (zero padded K)
// ---------------------------------------------------------------------------
__global__ void k_prep(const float* __restrict__ src, _Float16* __restrict__ dst,
                       int K, int N, int Kpad) {
  int idx = blockIdx.x * 256 + threadIdx.x;
  if (idx >= N * Kpad) return;
  int n = idx / Kpad, k = idx - n * Kpad;
  dst[idx] = (k < K) ? (_Float16)src[k * N + n] : (_Float16)0.f;
}

// ---------------------------------------------------------------------------
// Kernel 1: per-batch fused encoder + value MLP + landmark attention.
// grid = 16384 workgroups x 512 threads. Wave w owns output cols [16w,16w+16).
// ---------------------------------------------------------------------------
__global__ __launch_bounds__(512) void k_encoder(
    const float* __restrict__ obs,
    const _Float16* __restrict__ wt1,   // [256][64]  enc_w1^T, K padded 48->64
    const _Float16* __restrict__ wt2,   // [256][256] enc_w2^T
    const _Float16* __restrict__ wtv1,  // [256][256] val_w1^T
    const _Float16* __restrict__ wtv2,  // [256][256] val_w2^T
    const float* __restrict__ b1, const float* __restrict__ b2,
    const float* __restrict__ bv1, const float* __restrict__ bv2,
    float* __restrict__ omean_out) {    // [16384][256]
  __shared__ __align__(32) _Float16 xt[16][64];
  __shared__ __align__(32) _Float16 actA[16][HID];
  __shared__ __align__(32) _Float16 emb[16][HID];
  __shared__ __align__(32) _Float16 vv[16][HID];
  __shared__ float qv[HID];
  __shared__ float sc[16];

  const int b    = blockIdx.x;
  const int tid  = threadIdx.x;
  const int wave = tid >> 5;
  const int lane = tid & 31;
  const int m    = lane & 15;   // A row; column within 16x16 C tile
  const int hh   = lane >> 4;   // lane half
  const int ncol = wave * 16 + m;

  // Build x tile: x[s] = concat(obs[(16b+s) mod B, :32], obs[b, 96+16s : +16], pad)
  for (int idx = tid; idx < 16 * 64; idx += 512) {
    int s = idx >> 6, c = idx & 63;
    float v;
    if (c < 32)      v = obs[(((b << 4) + s) & (BATCH - 1)) * 480 + c];
    else if (c < 48) v = obs[b * 480 + 96 + (s << 4) + (c - 32)];
    else             v = 0.f;
    xt[s][c] = (_Float16)v;
  }
  __syncthreads();

  // ---- enc layer 1: [16x64] @ [64x16] ----
  {
    f8 c0 = {};
    #pragma unroll
    for (int k0 = 0; k0 < 64; k0 += 32)
      c0 = wmma_f16(load_a_lds(&xt[m][0], k0, hh), load_b_gl(wt1 + ncol * 64, k0, hh), c0);
    float bias = b1[ncol];
    #pragma unroll
    for (int r = 0; r < 8; ++r)
      actA[r + hh * 8][ncol] = (_Float16)ftanh(c0[r] + bias);
  }
  __syncthreads();

  // ---- enc layer 2 -> emb ----
  {
    f8 c0 = {};
    #pragma unroll
    for (int k0 = 0; k0 < HID; k0 += 32)
      c0 = wmma_f16(load_a_lds(&actA[m][0], k0, hh), load_b_gl(wt2 + ncol * HID, k0, hh), c0);
    float bias = b2[ncol];
    #pragma unroll
    for (int r = 0; r < 8; ++r)
      emb[r + hh * 8][ncol] = (_Float16)ftanh(c0[r] + bias);
  }
  __syncthreads();

  // ---- val layer 1: emb -> actA (reuse) ----
  {
    f8 c0 = {};
    #pragma unroll
    for (int k0 = 0; k0 < HID; k0 += 32)
      c0 = wmma_f16(load_a_lds(&emb[m][0], k0, hh), load_b_gl(wtv1 + ncol * HID, k0, hh), c0);
    float bias = bv1[ncol];
    #pragma unroll
    for (int r = 0; r < 8; ++r)
      actA[r + hh * 8][ncol] = (_Float16)ftanh(c0[r] + bias);
  }
  __syncthreads();

  // ---- val layer 2 -> vv ----
  {
    f8 c0 = {};
    #pragma unroll
    for (int k0 = 0; k0 < HID; k0 += 32)
      c0 = wmma_f16(load_a_lds(&actA[m][0], k0, hh), load_b_gl(wtv2 + ncol * HID, k0, hh), c0);
    float bias = bv2[ncol];
    #pragma unroll
    for (int r = 0; r < 8; ++r)
      vv[r + hh * 8][ncol] = (_Float16)ftanh(c0[r] + bias);
  }
  __syncthreads();

  // ---- attention over 16 landmark slots ----
  // q[j] = mean_s emb[s][j]
  for (int j = tid; j < HID; j += 512) {
    float s = 0.f;
    #pragma unroll
    for (int r = 0; r < 16; ++r) s += (float)emb[r][j];
    qv[j] = s * (1.f / 16.f);
  }
  __syncthreads();
  // wave w computes score for slot w (wave32 butterfly reduce)
  {
    float p = 0.f;
    for (int j = lane; j < HID; j += 32) p += qv[j] * (float)emb[wave][j];
    #pragma unroll
    for (int off = 16; off > 0; off >>= 1) p += __shfl_xor(p, off, 32);
    if (lane == 0) sc[wave] = p * (1.f / 16.f);  // / sqrt(H)
  }
  __syncthreads();
  // softmax(16) + weighted sum of value rows
  float mx = sc[0];
  #pragma unroll
  for (int s2 = 1; s2 < 16; ++s2) mx = fmaxf(mx, sc[s2]);
  float att[16], den = 0.f;
  #pragma unroll
  for (int s2 = 0; s2 < 16; ++s2) { att[s2] = __expf(sc[s2] - mx); den += att[s2]; }
  float inv = __builtin_amdgcn_rcpf(den);
  for (int j = tid; j < HID; j += 512) {
    float o = 0.f;
    #pragma unroll
    for (int s2 = 0; s2 < 16; ++s2) o += att[s2] * (float)vv[s2][j];
    omean_out[b * HID + j] = o * inv;
  }
}

// ---------------------------------------------------------------------------
// Kernel 2: fused vals-MLP + layernorm + grouped attention (+ x4 agent tile).
// grid = 1024 workgroups x 512 threads; WG handles omean rows [16*wg, +16)
// = groups [4*wg, +4) with P = 4 rows each.
// ---------------------------------------------------------------------------
__global__ __launch_bounds__(512) void k_tail(
    const float* __restrict__ omean,     // [16384][256]
    const _Float16* __restrict__ wts1,   // [256][256] vals_w1^T
    const _Float16* __restrict__ wts2,   // [256][256] vals_w2^T
    const float* __restrict__ bs1, const float* __restrict__ bs2,
    const float* __restrict__ ln_g, const float* __restrict__ ln_b,
    float* __restrict__ att_out) {       // [16384][256]
  __shared__ __align__(32) _Float16 A16[16][HID];
  __shared__ __align__(32) _Float16 act[16][HID];
  __shared__ float om32[16][HID];
  __shared__ float ov[16][HID];
  __shared__ float qg[4][HID];
  __shared__ float scg[4][8][4];

  const int r0   = blockIdx.x * 16;
  const int tid  = threadIdx.x;
  const int wave = tid >> 5;
  const int lane = tid & 31;
  const int m    = lane & 15;
  const int hh   = lane >> 4;
  const int ncol = wave * 16 + m;

  for (int idx = tid; idx < 16 * HID; idx += 512) {
    int r = idx >> 8, j = idx & 255;
    float v = omean[(r0 + r) * HID + j];
    om32[r][j] = v;
    A16[r][j]  = (_Float16)v;
  }
  __syncthreads();

  // ---- vals layer 1 ----
  {
    f8 c0 = {};
    #pragma unroll
    for (int k0 = 0; k0 < HID; k0 += 32)
      c0 = wmma_f16(load_a_lds(&A16[m][0], k0, hh), load_b_gl(wts1 + ncol * HID, k0, hh), c0);
    float bias = bs1[ncol];
    #pragma unroll
    for (int r = 0; r < 8; ++r)
      act[r + hh * 8][ncol] = (_Float16)ftanh(c0[r] + bias);
  }
  __syncthreads();

  // ---- vals layer 2 -> ov (f32 for layernorm) ----
  {
    f8 c0 = {};
    #pragma unroll
    for (int k0 = 0; k0 < HID; k0 += 32)
      c0 = wmma_f16(load_a_lds(&act[m][0], k0, hh), load_b_gl(wts2 + ncol * HID, k0, hh), c0);
    float bias = bs2[ncol];
    #pragma unroll
    for (int r = 0; r < 8; ++r)
      ov[r + hh * 8][ncol] = ftanh(c0[r] + bias);
  }
  __syncthreads();

  // ---- layernorm per row (wave r handles row r; eps = 1e-6) ----
  {
    float s = 0.f, s2 = 0.f;
    for (int j = lane; j < HID; j += 32) { float x = ov[wave][j]; s += x; s2 += x * x; }
    #pragma unroll
    for (int off = 16; off > 0; off >>= 1) { s += __shfl_xor(s, off, 32); s2 += __shfl_xor(s2, off, 32); }
    float mu   = s * (1.f / 256.f);
    float var  = s2 * (1.f / 256.f) - mu * mu;
    float rstd = rsqrtf(var + 1e-6f);
    for (int j = lane; j < HID; j += 32)
      ov[wave][j] = (ov[wave][j] - mu) * rstd * ln_g[j] + ln_b[j];
  }
  __syncthreads();

  // ---- grouped attention: 4 groups of P=4 rows, 8 heads x 32 dims ----
  for (int idx = tid; idx < 4 * HID; idx += 512) {
    int gi = idx >> 8, j = idx & 255;
    qg[gi][j] = 0.25f * (om32[gi * 4 + 0][j] + om32[gi * 4 + 1][j] +
                         om32[gi * 4 + 2][j] + om32[gi * 4 + 3][j]);
  }
  __syncthreads();
  if (tid < 128) {  // 4 groups x 8 heads x 4 slots
    int gi = tid >> 5, h = (tid >> 2) & 7, p = tid & 3;
    float s = 0.f;
    #pragma unroll
    for (int a2 = 0; a2 < 32; ++a2)
      s += qg[gi][h * 32 + a2] * om32[gi * 4 + p][h * 32 + a2];
    scg[gi][h][p] = s * 0.17677669529663687f;  // / sqrt(32)
  }
  __syncthreads();
  for (int idx = tid; idx < 4 * HID; idx += 512) {
    int gi = idx >> 8, j = idx & 255, h = j >> 5;
    float s0 = scg[gi][h][0], s1 = scg[gi][h][1], s2 = scg[gi][h][2], s3 = scg[gi][h][3];
    float mx = fmaxf(fmaxf(s0, s1), fmaxf(s2, s3));
    float e0 = __expf(s0 - mx), e1 = __expf(s1 - mx), e2 = __expf(s2 - mx), e3 = __expf(s3 - mx);
    float inv = __builtin_amdgcn_rcpf(e0 + e1 + e2 + e3);
    float o = (e0 * ov[gi * 4 + 0][j] + e1 * ov[gi * 4 + 1][j] +
               e2 * ov[gi * 4 + 2][j] + e3 * ov[gi * 4 + 3][j]) * inv;
    int g = blockIdx.x * 4 + gi;  // [0, 4096)
    att_out[(g        ) * HID + j] = o;   // tile(out, (NA=4, 1))
    att_out[(g +  4096) * HID + j] = o;
    att_out[(g +  8192) * HID + j] = o;
    att_out[(g + 12288) * HID + j] = o;
  }
}

// ---------------------------------------------------------------------------
extern "C" void kernel_launch(void* const* d_in, const int* in_sizes, int n_in,
                              void* d_out, int out_size, void* d_ws, size_t ws_size,
                              hipStream_t stream) {
  (void)in_sizes; (void)n_in; (void)out_size; (void)ws_size;
  const float* obs     = (const float*)d_in[0];
  const float* enc_w1  = (const float*)d_in[1];
  const float* enc_b1  = (const float*)d_in[2];
  const float* enc_w2  = (const float*)d_in[3];
  const float* enc_b2  = (const float*)d_in[4];
  const float* val_w1  = (const float*)d_in[5];
  const float* val_b1  = (const float*)d_in[6];
  const float* val_w2  = (const float*)d_in[7];
  const float* val_b2  = (const float*)d_in[8];
  const float* vals_w1 = (const float*)d_in[9];
  const float* vals_b1 = (const float*)d_in[10];
  const float* vals_w2 = (const float*)d_in[11];
  const float* vals_b2 = (const float*)d_in[12];
  const float* ln_g    = (const float*)d_in[13];
  const float* ln_b    = (const float*)d_in[14];

  // Workspace: transposed f16 weights (~672 KB total).
  _Float16* wt1  = (_Float16*)d_ws;       // 256*64
  _Float16* wt2  = wt1  + 256 * 64;       // 256*256
  _Float16* wtv1 = wt2  + 256 * 256;
  _Float16* wtv2 = wtv1 + 256 * 256;
  _Float16* wts1 = wtv2 + 256 * 256;
  _Float16* wts2 = wts1 + 256 * 256;

  k_prep<<<(256 * 64 + 255) / 256, 256, 0, stream>>>(enc_w1, wt1, 48, 256, 64);
  k_prep<<<256 * 256 / 256, 256, 0, stream>>>(enc_w2,  wt2,  256, 256, 256);
  k_prep<<<256 * 256 / 256, 256, 0, stream>>>(val_w1,  wtv1, 256, 256, 256);
  k_prep<<<256 * 256 / 256, 256, 0, stream>>>(val_w2,  wtv2, 256, 256, 256);
  k_prep<<<256 * 256 / 256, 256, 0, stream>>>(vals_w1, wts1, 256, 256, 256);
  k_prep<<<256 * 256 / 256, 256, 0, stream>>>(vals_w2, wts2, 256, 256, 256);

  float* att_out   = (float*)d_out;                  // [16384][256] region 1
  float* omean_out = (float*)d_out + BATCH * HID;    // [16384][256] region 2

  k_encoder<<<BATCH, 512, 0, stream>>>(obs, wt1, wt2, wtv1, wtv2,
                                       enc_b1, enc_b2, val_b1, val_b2, omean_out);
  k_tail<<<1024, 512, 0, stream>>>(omean_out, wts1, wts2, vals_b1, vals_b2,
                                   ln_g, ln_b, att_out);
}
